// GroupedQueryAttentionExercise_43215960932838
// MI455X (gfx1250) — matrix operand order repaired and checked by reading, not compiled, not run on hardware
//
#include <hip/hip_runtime.h>
#include <hip/hip_bf16.h>

#define D_MODEL 2048
#define N_HEADS 32
#define N_KV    8
#define N_REP   4
#define DK      64
#define KV_DIM  512
#define BSZ     4
#define SEQ     2048
#define MTOK    (BSZ*SEQ)

typedef _Float16 v4h  __attribute__((ext_vector_type(4)));
typedef _Float16 v8h  __attribute__((ext_vector_type(8)));
typedef _Float16 v16h __attribute__((ext_vector_type(16)));
typedef float    v8f  __attribute__((ext_vector_type(8)));
typedef int      v4i  __attribute__((vector_size(16)));

union HFrag { v16h v; v8h h[2]; };

__device__ __forceinline__ v8f wmma_f16(const HFrag& a, const HFrag& b, v8f c) {
  // D = A(16x32 f16) x B(32x16 f16) + C(16x16 f32)
  return __builtin_amdgcn_wmma_f32_16x16x32_f16(false, a.v, false, b.v,
                                                (short)0, c, false, false);
}

// ---------- 16-lane row reductions via DPP row_ror (no LDS, no waits) ----------
__device__ __forceinline__ float red_max16(float v) {
  int x;
  x = __float_as_int(v);
  v = fmaxf(v, __int_as_float(__builtin_amdgcn_update_dpp(x, x, 0x121, 0xf, 0xf, false)));
  x = __float_as_int(v);
  v = fmaxf(v, __int_as_float(__builtin_amdgcn_update_dpp(x, x, 0x122, 0xf, 0xf, false)));
  x = __float_as_int(v);
  v = fmaxf(v, __int_as_float(__builtin_amdgcn_update_dpp(x, x, 0x124, 0xf, 0xf, false)));
  x = __float_as_int(v);
  v = fmaxf(v, __int_as_float(__builtin_amdgcn_update_dpp(x, x, 0x128, 0xf, 0xf, false)));
  return v;
}
__device__ __forceinline__ float red_add16(float v) {
  int x;
  x = __float_as_int(v);
  v = v + __int_as_float(__builtin_amdgcn_update_dpp(x, x, 0x121, 0xf, 0xf, false));
  x = __float_as_int(v);
  v = v + __int_as_float(__builtin_amdgcn_update_dpp(x, x, 0x122, 0xf, 0xf, false));
  x = __float_as_int(v);
  v = v + __int_as_float(__builtin_amdgcn_update_dpp(x, x, 0x124, 0xf, 0xf, false));
  x = __float_as_int(v);
  v = v + __int_as_float(__builtin_amdgcn_update_dpp(x, x, 0x128, 0xf, 0xf, false));
  return v;
}

// ---------- async global->LDS copy (gfx1250 path, guarded) ----------
#if defined(__has_builtin)
#if __has_builtin(__builtin_amdgcn_global_load_async_to_lds_b128)
#define HAVE_ASYNC_LDS 1
#endif
#endif

__device__ __forceinline__ void copy16_g2l(void* lds, const _Float16* g) {
#if defined(HAVE_ASYNC_LDS)
  __builtin_amdgcn_global_load_async_to_lds_b128(
      (__attribute__((address_space(1))) v4i*)g,
      (__attribute__((address_space(3))) v4i*)lds, 0, 0);
#else
  *(v8h*)lds = *(const v8h*)g;
#endif
}
__device__ __forceinline__ void async_join() {
#if defined(HAVE_ASYNC_LDS)
#if __has_builtin(__builtin_amdgcn_s_wait_asynccnt)
  __builtin_amdgcn_s_wait_asynccnt(0);
#else
  asm volatile("s_wait_asynccnt 0x0" ::: "memory");
#endif
#endif
}

// ---------------- elementwise f32 -> f16 ----------------
__global__ void convert_f32_f16(const float* __restrict__ in,
                                _Float16* __restrict__ out, int n) {
  int i = (blockIdx.x * blockDim.x + threadIdx.x) * 4;
  if (i < n) {
    float4 f = *(const float4*)(in + i);
    v4h o = { (_Float16)f.x, (_Float16)f.y, (_Float16)f.z, (_Float16)f.w };
    *(v4h*)(out + i) = o;
  }
}

// ---------------- weight transpose: W f32 [K][N] -> Wt f16 [N][K] ----------------
__global__ void transpose_w_f16(const float* __restrict__ W,
                                _Float16* __restrict__ Wt, int K, int N) {
  __shared__ _Float16 sh[32][33];
  int n0 = blockIdx.x * 32, k0 = blockIdx.y * 32;
  int tx = threadIdx.x, ty = threadIdx.y;
#pragma unroll
  for (int i = 0; i < 4; ++i) {
    int k = k0 + ty + i * 8;
    sh[ty + i * 8][tx] = (_Float16)W[(size_t)k * N + n0 + tx];
  }
  __syncthreads();
#pragma unroll
  for (int i = 0; i < 4; ++i) {
    int n = n0 + ty + i * 8;
    Wt[(size_t)n * K + k0 + tx] = sh[tx][ty + i * 8];
  }
}

// ---------------- V transpose: Vh [B*T][KV_DIM] -> Vt [B*NKV][DK][T] ----------------
__global__ void transpose_v(const _Float16* __restrict__ V,
                            _Float16* __restrict__ Vt) {
  __shared__ _Float16 sh[32][33];
  int t0 = blockIdx.x * 32, d0 = blockIdx.y * 32, bg = blockIdx.z;
  int b = bg / N_KV, g = bg % N_KV;
  int tx = threadIdx.x, ty = threadIdx.y;
#pragma unroll
  for (int i = 0; i < 4; ++i) {
    int t = t0 + ty + i * 8;
    sh[ty + i * 8][tx] = V[((size_t)(b * SEQ + t)) * KV_DIM + g * DK + d0 + tx];
  }
  __syncthreads();
#pragma unroll
  for (int i = 0; i < 4; ++i) {
    int d = d0 + ty + i * 8;
    Vt[((size_t)(bg * DK + d)) * SEQ + t0 + tx] = sh[tx][ty + i * 8];
  }
}

// ---------------- WMMA GEMM: C = A[MxK] * Bt[NxK]^T + bias ----------------
// block = 256 threads = 8 waves; block tile 128(M) x 64(N); wave tile 16x64.
// B tile for a 128-deep k-chunk staged once in LDS (async), shared by 8 waves.
// grid = (N/64, M/128).
template <typename OutT>
__global__ void __launch_bounds__(256)
gemm_bias_wmma(const _Float16* __restrict__ A, const _Float16* __restrict__ Bt,
               const float* __restrict__ bias, OutT* __restrict__ C,
               int M, int N, int K) {
  __shared__ __align__(16) _Float16 BT[64][136];  // 64 cols x 128 k (+pad)
  int wave = threadIdx.x >> 5, lane = threadIdx.x & 31;
  int r = lane & 15, hh = lane >> 4;
  int m0 = blockIdx.y * 128 + wave * 16;
  int n0 = blockIdx.x * 64;
  const _Float16* Ar = A + (size_t)(m0 + r) * K;  // lane r = row of A tile
  v8f acc[4] = {{}, {}, {}, {}};

  for (int k0 = 0; k0 < K; k0 += 128) {
    // stage B chunk: 64 rows x 128 halfs = 1024 x 16B; 4 chunks per thread
#pragma unroll
    for (int i = 0; i < 4; ++i) {
      int c = threadIdx.x * 4 + i;  // 0..1023
      int row = c >> 4, seg = c & 15;
      copy16_g2l(&BT[row][seg * 8], Bt + (size_t)(n0 + row) * K + k0 + seg * 8);
    }
    if (k0 + 128 < K) __builtin_prefetch(Ar + k0 + 128, 0, 1);
    async_join();
    __syncthreads();
#pragma unroll
    for (int kk = 0; kk < 128; kk += 32) {
      HFrag a;
      a.h[0] = *(const v8h*)(Ar + k0 + kk + hh * 8);
      a.h[1] = *(const v8h*)(Ar + k0 + kk + 16 + hh * 8);
#pragma unroll
      for (int t = 0; t < 4; ++t) {
        HFrag b;
        b.h[0] = *(const v8h*)&BT[t * 16 + r][kk + hh * 8];
        b.h[1] = *(const v8h*)&BT[t * 16 + r][kk + 16 + hh * 8];
        acc[t] = wmma_f16(a, b, acc[t]);
      }
    }
    __syncthreads();
  }
#pragma unroll
  for (int t = 0; t < 4; ++t) {
    float bvv = bias[n0 + t * 16 + r];
#pragma unroll
    for (int g = 0; g < 8; ++g) {
      int m = m0 + g + 8 * hh;  // D layout: vgpr g, half hh -> row g+8*hh
      C[(size_t)m * N + n0 + t * 16 + r] = (OutT)(acc[t][g] + bvv);
    }
  }
}

// ---------------- flash attention block step ----------------
// MASK=false: fully-unmasked interior block (no compares/selects).
template <bool MASK>
__device__ __forceinline__ void attn_block(
    int j0, int q0, int r, int hh, const HFrag& aq0, const HFrag& aq1,
    const _Float16* __restrict__ Kbase, const _Float16* __restrict__ Vbase,
    v8f (&o)[4], float (&rmax)[8], float (&rsum)[8], _Float16 (&Pw)[16][48]) {
  v8f s0 = {}, s1 = {};
  const _Float16* Kr0 = Kbase + (size_t)(j0 + r) * KV_DIM;
  const _Float16* Kr1 = Kr0 + (size_t)16 * KV_DIM;
  HFrag bk;
  bk.h[0] = *(const v8h*)(Kr0 + hh * 16);
  bk.h[1] = *(const v8h*)(Kr0 + hh * 16 + 8);
  s0 = wmma_f16(aq0, bk, s0);
  bk.h[0] = *(const v8h*)(Kr0 + 32 + hh * 16);
  bk.h[1] = *(const v8h*)(Kr0 + 40 + hh * 16);
  s0 = wmma_f16(aq1, bk, s0);
  bk.h[0] = *(const v8h*)(Kr1 + hh * 16);
  bk.h[1] = *(const v8h*)(Kr1 + hh * 16 + 8);
  s1 = wmma_f16(aq0, bk, s1);
  bk.h[0] = *(const v8h*)(Kr1 + 32 + hh * 16);
  bk.h[1] = *(const v8h*)(Kr1 + 40 + hh * 16);
  s1 = wmma_f16(aq1, bk, s1);

  // online softmax: rows live in one 16-lane half, one row per acc vgpr
  int ki0 = j0 + r, ki1 = ki0 + 16;
  int qbase = q0 + 8 * hh;
#pragma unroll
  for (int g = 0; g < 8; ++g) {
    float v0 = s0[g] * 0.125f;
    float v1 = s1[g] * 0.125f;
    if (MASK) {
      int qi = qbase + g;
      v0 = (ki0 <= qi) ? v0 : -1e30f;
      v1 = (ki1 <= qi) ? v1 : -1e30f;
    }
    float mx = red_max16(fmaxf(v0, v1));
    float nm = fmaxf(rmax[g], mx);
    float corr = __expf(rmax[g] - nm);
    rmax[g] = nm;
    float e0 = __expf(v0 - nm);
    float e1 = __expf(v1 - nm);
    float ps = red_add16(e0 + e1);
    rsum[g] = rsum[g] * corr + ps;
    o[0][g] *= corr; o[1][g] *= corr; o[2][g] *= corr; o[3][g] *= corr;
    Pw[g + 8 * hh][r] = (_Float16)e0;
    Pw[g + 8 * hh][16 + r] = (_Float16)e1;
  }
  asm volatile("s_wait_dscnt 0x0" ::: "memory");
  HFrag ap;  // P as 16x32 A fragment
  ap.h[0] = *(const v8h*)&Pw[r][hh * 8];
  ap.h[1] = *(const v8h*)&Pw[r][16 + hh * 8];

#pragma unroll
  for (int t = 0; t < 4; ++t) {
    const _Float16* Vr = Vbase + (size_t)(t * 16 + r) * SEQ + j0;
    HFrag bv_;
    bv_.h[0] = *(const v8h*)(Vr + hh * 8);
    bv_.h[1] = *(const v8h*)(Vr + 16 + hh * 8);
    o[t] = wmma_f16(ap, bv_, o[t]);
  }
}

// ---------------- flash attention ----------------
// grid = (T/64, N_HEADS, B); block = 128 = 4 waves; each wave: 16 query rows.
__global__ void __launch_bounds__(128)
attn_fwd(const _Float16* __restrict__ Qh, const _Float16* __restrict__ Kh,
         const _Float16* __restrict__ Vt, _Float16* __restrict__ Oh) {
  int b = blockIdx.z, head = blockIdx.y;
  int kvh = head >> 2;  // N_REP = 4
  int wave = threadIdx.x >> 5, lane = threadIdx.x & 31;
  int r = lane & 15, hh = lane >> 4;
  int q0 = blockIdx.x * 64 + wave * 16;
  __shared__ __align__(16) _Float16 Pst[4][16][48];

  const _Float16* Qr = Qh + ((size_t)(b * SEQ + q0 + r)) * D_MODEL + head * DK;
  HFrag aq0, aq1;
  aq0.h[0] = *(const v8h*)(Qr + hh * 16);
  aq0.h[1] = *(const v8h*)(Qr + hh * 16 + 8);
  aq1.h[0] = *(const v8h*)(Qr + 32 + hh * 16);
  aq1.h[1] = *(const v8h*)(Qr + 40 + hh * 16);

  v8f o[4] = {{}, {}, {}, {}};
  float rmax[8], rsum[8];
#pragma unroll
  for (int g = 0; g < 8; ++g) { rmax[g] = -1e30f; rsum[g] = 0.f; }

  const _Float16* Kbase = Kh + (size_t)b * SEQ * KV_DIM + kvh * DK;
  const _Float16* Vbase = Vt + ((size_t)(b * N_KV + kvh)) * DK * SEQ;

  int nfull = (q0 + 1) >> 5;   // blocks with j0+31 <= q0: no masking needed
  int nblk  = (q0 + 47) >> 5;  // total 32-key blocks (keys <= q0+15)
  int jb = 0;
  for (; jb < nfull; ++jb)
    attn_block<false>(jb << 5, q0, r, hh, aq0, aq1, Kbase, Vbase,
                      o, rmax, rsum, Pst[wave]);
  for (; jb < nblk; ++jb)
    attn_block<true>(jb << 5, q0, r, hh, aq0, aq1, Kbase, Vbase,
                     o, rmax, rsum, Pst[wave]);

#pragma unroll
  for (int g = 0; g < 8; ++g) {
    float inv = 1.0f / rsum[g];
    size_t row = (size_t)(b * SEQ + q0 + g + 8 * hh);
    _Float16* dst = Oh + row * D_MODEL + head * DK + r;
    dst[0]  = (_Float16)(o[0][g] * inv);
    dst[16] = (_Float16)(o[1][g] * inv);
    dst[32] = (_Float16)(o[2][g] * inv);
    dst[48] = (_Float16)(o[3][g] * inv);
  }
}

extern "C" void kernel_launch(void* const* d_in, const int* in_sizes, int n_in,
                              void* d_out, int out_size, void* d_ws, size_t ws_size,
                              hipStream_t stream) {
  const float* x  = (const float*)d_in[0];
  const float* Wq = (const float*)d_in[1];
  const float* bq = (const float*)d_in[2];
  const float* Wk = (const float*)d_in[3];
  const float* bk = (const float*)d_in[4];
  const float* Wv = (const float*)d_in[5];
  const float* bv = (const float*)d_in[6];
  const float* Wo = (const float*)d_in[7];
  const float* bo = (const float*)d_in[8];
  float* out = (float*)d_out;

  char* ws = (char*)d_ws;
  size_t off = 0;
  auto alloc = [&](size_t bytes) -> void* {
    void* p = ws + off;
    off += (bytes + 255) & ~(size_t)255;
    return p;
  };
  _Float16* xh  = (_Float16*)alloc((size_t)MTOK * D_MODEL * 2);
  _Float16* Wqt = (_Float16*)alloc((size_t)D_MODEL * D_MODEL * 2);
  _Float16* Wkt = (_Float16*)alloc((size_t)KV_DIM * D_MODEL * 2);
  _Float16* Wvt = (_Float16*)alloc((size_t)KV_DIM * D_MODEL * 2);
  _Float16* Wot = (_Float16*)alloc((size_t)D_MODEL * D_MODEL * 2);
  _Float16* Qhp = (_Float16*)alloc((size_t)MTOK * D_MODEL * 2);
  _Float16* Khp = (_Float16*)alloc((size_t)MTOK * KV_DIM * 2);
  _Float16* Vhp = (_Float16*)alloc((size_t)MTOK * KV_DIM * 2);
  _Float16* Vtp = (_Float16*)alloc((size_t)MTOK * KV_DIM * 2);
  _Float16* Ohp = (_Float16*)alloc((size_t)MTOK * D_MODEL * 2);

  {
    int n = MTOK * D_MODEL;
    convert_f32_f16<<<n / 4 / 256, 256, 0, stream>>>(x, xh, n);
  }
  transpose_w_f16<<<dim3(D_MODEL / 32, D_MODEL / 32), dim3(32, 8), 0, stream>>>(
      Wq, Wqt, D_MODEL, D_MODEL);
  transpose_w_f16<<<dim3(KV_DIM / 32, D_MODEL / 32), dim3(32, 8), 0, stream>>>(
      Wk, Wkt, D_MODEL, KV_DIM);
  transpose_w_f16<<<dim3(KV_DIM / 32, D_MODEL / 32), dim3(32, 8), 0, stream>>>(
      Wv, Wvt, D_MODEL, KV_DIM);
  transpose_w_f16<<<dim3(D_MODEL / 32, D_MODEL / 32), dim3(32, 8), 0, stream>>>(
      Wo, Wot, D_MODEL, D_MODEL);

  gemm_bias_wmma<_Float16><<<dim3(D_MODEL / 64, MTOK / 128), 256, 0, stream>>>(
      xh, Wqt, bq, Qhp, MTOK, D_MODEL, D_MODEL);
  gemm_bias_wmma<_Float16><<<dim3(KV_DIM / 64, MTOK / 128), 256, 0, stream>>>(
      xh, Wkt, bk, Khp, MTOK, KV_DIM, D_MODEL);
  gemm_bias_wmma<_Float16><<<dim3(KV_DIM / 64, MTOK / 128), 256, 0, stream>>>(
      xh, Wvt, bv, Vhp, MTOK, KV_DIM, D_MODEL);

  transpose_v<<<dim3(SEQ / 32, DK / 32, BSZ * N_KV), dim3(32, 8), 0, stream>>>(
      Vhp, Vtp);

  attn_fwd<<<dim3(SEQ / 64, N_HEADS, BSZ), 128, 0, stream>>>(Qhp, Khp, Vtp, Ohp);

  gemm_bias_wmma<float><<<dim3(D_MODEL / 64, MTOK / 128), 256, 0, stream>>>(
      Ohp, Wot, bo, out, MTOK, D_MODEL, D_MODEL);
}
